// MultiHeadAttention_45329084842341
// MI455X (gfx1250) — matrix operand order repaired and checked by reading, not compiled, not run on hardware
//
#include <hip/hip_runtime.h>

// ---------------------------------------------------------------------------
// Types for CDNA5 WMMA (wave32): v16bf = A/B fragment, v8f = C/D fragment
// ---------------------------------------------------------------------------
typedef __attribute__((ext_vector_type(16))) __bf16 v16bf;
typedef __attribute__((ext_vector_type(8)))  __bf16 v8bf;
typedef __attribute__((ext_vector_type(8)))  float  v8f;

#ifndef __has_builtin
#define __has_builtin(x) 0
#endif

// Async global->LDS copy (CDNA5 GLOBAL_LOAD_ASYNC_TO_LDS_B128, ASYNCcnt-tracked)
#if __has_builtin(__builtin_amdgcn_global_load_async_to_lds_b128)
#define USE_ASYNC_LDS 1
#else
#define USE_ASYNC_LDS 0
#endif

#if __has_builtin(__builtin_amdgcn_s_wait_asynccnt)
#define WAIT_ASYNC(n) __builtin_amdgcn_s_wait_asynccnt(n)
#elif USE_ASYNC_LDS
#define WAIT_ASYNC(n) asm volatile("s_wait_asynccnt %0" ::"i"(n) : "memory")
#else
#define WAIT_ASYNC(n) ((void)0)
#endif

#if USE_ASYNC_LDS
// Exact parameter types per hipcc diagnostic:
//   arg0: int __attribute__((vector_size(16))) __device__(AS1) *
//   arg1: LDS-side pointer (AS3), same pointee
typedef int v4i_gcc __attribute__((vector_size(4 * sizeof(int))));
typedef __attribute__((address_space(1))) v4i_gcc* as1_v4i;
typedef __attribute__((address_space(3))) v4i_gcc* as3_v4i;
#endif

__device__ __forceinline__ float bf2f(__bf16 h) {
  unsigned short u = __builtin_bit_cast(unsigned short, h);
  unsigned int x = ((unsigned int)u) << 16;
  return __builtin_bit_cast(float, x);
}
__device__ __forceinline__ __bf16 f2bf(float f) {
  unsigned int x = __builtin_bit_cast(unsigned int, f);
  x += 0x7FFFu + ((x >> 16) & 1u);  // round-to-nearest-even
  unsigned short u = (unsigned short)(x >> 16);
  return __builtin_bit_cast(__bf16, u);
}

// 16-byte global -> LDS copy: async DMA when available, else via VGPRs
__device__ __forceinline__ void cp16(__bf16* l, const __bf16* g) {
#if USE_ASYNC_LDS
  __builtin_amdgcn_global_load_async_to_lds_b128((as1_v4i)(g), (as3_v4i)(l), 0, 0);
#else
  *(v8bf*)l = *(const v8bf*)g;
#endif
}

// ---------------------------------------------------------------------------
// f32 -> bf16 conversion (plain)
// ---------------------------------------------------------------------------
__global__ void f32_to_bf16(const float* __restrict__ src, __bf16* __restrict__ dst,
                            long long n) {
  long long i = (long long)blockIdx.x * blockDim.x + threadIdx.x;
  if (i < n) dst[i] = f2bf(src[i]);
}

// ---------------------------------------------------------------------------
// f32 [R][C] -> bf16 [C][R] transposed conversion (32x32 LDS tile)
// ---------------------------------------------------------------------------
__global__ __launch_bounds__(256) void f32_to_bf16_T(const float* __restrict__ src,
                                                     __bf16* __restrict__ dst,
                                                     int R, int Cc) {
  __shared__ float t[32][33];
  const int bx = blockIdx.x * 32;  // source col base
  const int by = blockIdx.y * 32;  // source row base
  const int tx = threadIdx.x;      // 0..31
  const int ty = threadIdx.y;      // 0..7
#pragma unroll
  for (int i = ty; i < 32; i += 8)
    t[i][tx] = src[(long long)(by + i) * Cc + (bx + tx)];
  __syncthreads();
#pragma unroll
  for (int i = ty; i < 32; i += 8)
    dst[(long long)(bx + i) * R + (by + tx)] = f2bf(t[tx][i]);
}

// ---------------------------------------------------------------------------
// bf16 batched transpose: per batch z, src [R][C] -> dst [C][R]
// ---------------------------------------------------------------------------
__global__ __launch_bounds__(256) void bf16_T_batched(const __bf16* __restrict__ src,
                                                      __bf16* __restrict__ dst,
                                                      int R, int Cc,
                                                      long long sSrc, long long sDst) {
  __shared__ __bf16 t[32][33];
  const __bf16* s = src + (long long)blockIdx.z * sSrc;
  __bf16* d = dst + (long long)blockIdx.z * sDst;
  const int bx = blockIdx.x * 32;
  const int by = blockIdx.y * 32;
  const int tx = threadIdx.x;
  const int ty = threadIdx.y;
#pragma unroll
  for (int i = ty; i < 32; i += 8)
    t[i][tx] = s[(long long)(by + i) * Cc + (bx + tx)];
  __syncthreads();
#pragma unroll
  for (int i = ty; i < 32; i += 8)
    d[(long long)(bx + i) * R + (by + tx)] = t[tx][i];
}

// ---------------------------------------------------------------------------
// Batched bf16 WMMA GEMM:  C[b,h] = scale * (A[b,h] @ B[b,h]^T) (+bias)(+mask)(+res)
// A: [M][K] row-major; B: [N][K] row-major (K contiguous for both operands).
// Block = 256 threads = 8 waves; tile 128x128, K-step 32, double-buffered LDS
// with async global->LDS staging pipelined against WMMA.
// Waves arranged 4 (M) x 2 (N); each wave computes 32x64 via 2x4 WMMA tiles.
// ---------------------------------------------------------------------------
template <bool OUT_F32>
__global__ __launch_bounds__(256) void gemm_bf16_wmma(
    const __bf16* __restrict__ A, long long sAb, long long sAh, int lda,
    const __bf16* __restrict__ B, long long sBb, long long sBh, int ldb,
    void* __restrict__ C, long long sCb, long long sCh, int ldc,
    const float* __restrict__ bias,                                  // [N] or null
    const float* __restrict__ residual, long long sRb, int ldr,      // or null
    const unsigned char* __restrict__ mask, long long sMb, int ldm,  // or null
    float scale, int K, int nH) {
  __shared__ __bf16 Al[2][128][40];  // [buf][m][k], padded
  __shared__ __bf16 Bl[2][128][40];  // [buf][n][k], padded

  const int tid  = threadIdx.x;
  const int lane = tid & 31;
  const int wid  = tid >> 5;
  const int wm   = wid & 3;   // wave row 0..3
  const int wn   = wid >> 2;  // wave col 0..1
  const int lane_r  = lane & 15;
  const int lane_hi = lane >> 4;

  const int z = blockIdx.z;
  const int b = z / nH;
  const int h = z % nH;

  const __bf16* Ab = A + b * sAb + h * sAh;
  const __bf16* Bb = B + b * sBb + h * sBh;

  const int m0 = blockIdx.y * 128;
  const int n0 = blockIdx.x * 128;

  const int crow  = tid & 127;  // staging row index
  const int chalf = tid >> 7;   // which 16-wide K half

  const v8f vzero = {0.f, 0.f, 0.f, 0.f, 0.f, 0.f, 0.f, 0.f};
  v8f acc[2][4];
#pragma unroll
  for (int t = 0; t < 2; ++t)
#pragma unroll
    for (int c = 0; c < 4; ++c) acc[t][c] = vzero;

  // Stage one 128x32 A tile + 128x32 B tile into LDS buffer `buf`
  // (4 x 16-byte copies per thread -> ASYNCcnt += 4 on the async path).
  auto stage = [&](int k0, int buf) {
    const __bf16* asrc = Ab + (long long)(m0 + crow) * lda + (k0 + chalf * 16);
    __bf16* adst = &Al[buf][crow][chalf * 16];
    cp16(adst, asrc);
    cp16(adst + 8, asrc + 8);
    const __bf16* bsrc = Bb + (long long)(n0 + crow) * ldb + (k0 + chalf * 16);
    __bf16* bdst = &Bl[buf][crow][chalf * 16];
    cp16(bdst, bsrc);
    cp16(bdst + 8, bsrc + 8);
#if !USE_ASYNC_LDS
    if (k0 + 32 < K) {
      __builtin_prefetch(asrc + 32, 0, 3);  // global_prefetch_b8, near locality
      __builtin_prefetch(bsrc + 32, 0, 3);
    }
#endif
  };

  stage(0, 0);
  int p = 0;
  for (int k0 = 0; k0 < K; k0 += 32) {
    // Software pipeline: kick off next tile's async copies, then retire
    // the current tile's 4 in-order copies before consuming it.
    if (k0 + 32 < K) {
      stage(k0 + 32, p ^ 1);
      WAIT_ASYNC(4);
    } else {
      WAIT_ASYNC(0);
    }
    __syncthreads();

#pragma unroll
    for (int t = 0; t < 2; ++t) {
      // A fragment (16x32): lanes 0-15 hold K 0-7 & 16-23, lanes 16-31 hold 8-15 & 24-31
      const int ar  = wm * 32 + t * 16 + lane_r;
      const int akb = lane_hi * 8;
      union { v16bf v; v8bf q[2]; } af;
      af.q[0] = *(const v8bf*)&Al[p][ar][akb];
      af.q[1] = *(const v8bf*)&Al[p][ar][akb + 16];
#pragma unroll
      for (int c = 0; c < 4; ++c) {
        // B fragment (32x16): lane holds column, K = lane_hi*8 + 0..7 & 16 + lane_hi*8 + 0..7
        const int bc  = wn * 64 + c * 16 + lane_r;
        const int bkb = lane_hi * 8;
        union { v16bf v; v8bf q[2]; } bf;
        bf.q[0] = *(const v8bf*)&Bl[p][bc][bkb];
        bf.q[1] = *(const v8bf*)&Bl[p][bc][bkb + 16];
        acc[t][c] = __builtin_amdgcn_wmma_f32_16x16x32_bf16(
            false, af.v, false, bf.v, (short)0, acc[t][c], false, false);
      }
    }
    __syncthreads();
    p ^= 1;
  }

  // ---- epilogue ----
  const unsigned char* mb = mask ? (mask + b * sMb) : nullptr;
  const float* rb = residual ? (residual + b * sRb) : nullptr;
#pragma unroll
  for (int t = 0; t < 2; ++t) {
#pragma unroll
    for (int c = 0; c < 4; ++c) {
#pragma unroll
      for (int i = 0; i < 8; ++i) {
        // C/D layout: VGPR i -> row (lane_hi*8 + i), col = lane_r within 16x16 tile
        const int row = m0 + wm * 32 + t * 16 + lane_hi * 8 + i;
        const int col = n0 + wn * 64 + c * 16 + lane_r;
        float v = acc[t][c][i] * scale;
        if (bias) v += bias[col];
        if (mb && mb[(long long)row * ldm + col]) v = -1e9f;
        if (rb) v += rb[(long long)row * ldr + col];
        const long long coff = b * sCb + h * sCh + (long long)row * ldc + col;
        if (OUT_F32)
          ((float*)C)[coff] = v;
        else
          ((__bf16*)C)[coff] = f2bf(v);
      }
    }
  }
}

// ---------------------------------------------------------------------------
// Row softmax over bf16 scores (S = 1024, 256 threads, 4 elems/thread), in place
// ---------------------------------------------------------------------------
__global__ __launch_bounds__(256) void softmax_rows(__bf16* __restrict__ P, int S) {
  __shared__ float red[256];
  __bf16* p = P + (long long)blockIdx.x * S;
  const int tid = threadIdx.x;
  float vals[4];
  float mx = -3.0e38f;
#pragma unroll
  for (int i = 0; i < 4; ++i) {
    vals[i] = bf2f(p[tid + i * 256]);
    mx = fmaxf(mx, vals[i]);
  }
  red[tid] = mx;
  __syncthreads();
  for (int s = 128; s > 0; s >>= 1) {
    if (tid < s) red[tid] = fmaxf(red[tid], red[tid + s]);
    __syncthreads();
  }
  mx = red[0];
  __syncthreads();
  float sum = 0.f;
#pragma unroll
  for (int i = 0; i < 4; ++i) {
    vals[i] = __expf(vals[i] - mx);
    sum += vals[i];
  }
  red[tid] = sum;
  __syncthreads();
  for (int s = 128; s > 0; s >>= 1) {
    if (tid < s) red[tid] += red[tid + s];
    __syncthreads();
  }
  const float inv = 1.0f / red[0];
#pragma unroll
  for (int i = 0; i < 4; ++i) p[tid + i * 256] = f2bf(vals[i] * inv);
}

// ---------------------------------------------------------------------------
// LayerNorm over D=512 per row, in place (256 threads, 2 elems/thread)
// ---------------------------------------------------------------------------
__global__ __launch_bounds__(256) void layernorm_rows(float* __restrict__ X,
                                                      const float* __restrict__ gamma,
                                                      const float* __restrict__ beta,
                                                      int Dn) {
  __shared__ float r1[256];
  __shared__ float r2[256];
  float* x = X + (long long)blockIdx.x * Dn;
  const int tid = threadIdx.x;
  const float a0 = x[tid], a1 = x[tid + 256];
  r1[tid] = a0 + a1;
  r2[tid] = a0 * a0 + a1 * a1;
  __syncthreads();
  for (int s = 128; s > 0; s >>= 1) {
    if (tid < s) { r1[tid] += r1[tid + s]; r2[tid] += r2[tid + s]; }
    __syncthreads();
  }
  const float mu  = r1[0] / Dn;
  const float var = r2[0] / Dn - mu * mu;
  const float inv = rsqrtf(var + 1e-5f);
  x[tid]       = (a0 - mu) * inv * gamma[tid] + beta[tid];
  x[tid + 256] = (a1 - mu) * inv * gamma[tid + 256] + beta[tid + 256];
}

// ---------------------------------------------------------------------------
// Host-side launch
// ---------------------------------------------------------------------------
extern "C" void kernel_launch(void* const* d_in, const int* in_sizes, int n_in,
                              void* d_out, int out_size, void* d_ws, size_t ws_size,
                              hipStream_t stream) {
  (void)in_sizes; (void)n_in; (void)out_size; (void)ws_size;

  const float*         Qin   = (const float*)d_in[0];
  const float*         Kin   = (const float*)d_in[1];
  const float*         Vin   = (const float*)d_in[2];
  const unsigned char* mask  = (const unsigned char*)d_in[3];  // bool [B,S,S]
  const float*         Wq    = (const float*)d_in[4];
  const float*         bq    = (const float*)d_in[5];
  const float*         Wk    = (const float*)d_in[6];
  const float*         bk    = (const float*)d_in[7];
  const float*         Wv    = (const float*)d_in[8];
  const float*         bv    = (const float*)d_in[9];
  const float*         Wo    = (const float*)d_in[10];
  const float*         bo    = (const float*)d_in[11];
  const float*         gamma = (const float*)d_in[12];
  const float*         beta  = (const float*)d_in[13];

  constexpr int B = 8, S = 1024, D = 512, H = 8;
  constexpr int DH = 2 * D;   // 1024 per-head dim
  constexpr int HD = H * DH;  // 8192
  const long long BS = (long long)B * S;  // 8192 rows

  // ---- workspace layout (bf16 elements, 256B aligned chunks) ----
  __bf16* ws = (__bf16*)d_ws;
  long long off = 0;
  auto take = [&](long long elems) {
    __bf16* pp = ws + off;
    off += (elems + 127) & ~127LL;
    return pp;
  };
  __bf16* Qb  = take(BS * D);
  __bf16* Kb  = take(BS * D);
  __bf16* Vb  = take(BS * D);
  __bf16* WqT = take((long long)D * HD);  // [HD][D]
  __bf16* WkT = take((long long)D * HD);  // [HD][D]
  __bf16* WvT = take((long long)D * HD);  // [HD][D]
  __bf16* WoT = take((long long)HD * D);  // [D][HD]
  __bf16* qp  = take(BS * HD);            // [B*S][HD]
  __bf16* kp  = take(BS * HD);            // [B*S][HD]
  __bf16* vp  = take(BS * HD);            // [B*S][HD]
  __bf16* Pp  = take((long long)B * H * S * S);
  __bf16* ctx = take(BS * HD);
  // vpT [b][h][d][s] aliases qp: qp is dead after the scores GEMM,
  // and the v-transpose launches strictly after it on the same stream.
  __bf16* vpT = qp;

  const dim3 blk(256);
  const dim3 tblk(32, 8);

  // ---- 1) f32 -> bf16 conversions (weights transposed to [N][K]) ----
  {
    const long long nX = BS * D;
    const int gX = (int)((nX + 255) / 256);
    f32_to_bf16<<<gX, blk, 0, stream>>>(Qin, Qb, nX);
    f32_to_bf16<<<gX, blk, 0, stream>>>(Kin, Kb, nX);
    f32_to_bf16<<<gX, blk, 0, stream>>>(Vin, Vb, nX);
    dim3 gw(HD / 32, D / 32, 1);  // src [D][HD] -> dst [HD][D]
    f32_to_bf16_T<<<gw, tblk, 0, stream>>>(Wq, WqT, D, HD);
    f32_to_bf16_T<<<gw, tblk, 0, stream>>>(Wk, WkT, D, HD);
    f32_to_bf16_T<<<gw, tblk, 0, stream>>>(Wv, WvT, D, HD);
    dim3 go(D / 32, HD / 32, 1);  // src [HD][D] -> dst [D][HD]
    f32_to_bf16_T<<<go, tblk, 0, stream>>>(Wo, WoT, HD, D);
  }

  // ---- 2) QKV projections: [8192x512] @ [512x8192] + bias -> bf16 ----
  {
    dim3 g(HD / 128, (int)(BS / 128), 1);
    gemm_bf16_wmma<false><<<g, blk, 0, stream>>>(
        Qb, 0, 0, D, WqT, 0, 0, D, qp, 0, 0, HD,
        bq, nullptr, 0, 0, nullptr, 0, 0, 1.0f, D, 1);
    gemm_bf16_wmma<false><<<g, blk, 0, stream>>>(
        Kb, 0, 0, D, WkT, 0, 0, D, kp, 0, 0, HD,
        bk, nullptr, 0, 0, nullptr, 0, 0, 1.0f, D, 1);
    gemm_bf16_wmma<false><<<g, blk, 0, stream>>>(
        Vb, 0, 0, D, WvT, 0, 0, D, vp, 0, 0, HD,
        bv, nullptr, 0, 0, nullptr, 0, 0, 1.0f, D, 1);
  }

  // ---- 3) scores = (q @ k^T) / 32, masked -> bf16 P [B,H,S,S] ----
  {
    dim3 g(S / 128, S / 128, B * H);
    gemm_bf16_wmma<false><<<g, blk, 0, stream>>>(
        qp, (long long)S * HD, DH, HD,
        kp, (long long)S * HD, DH, HD,
        Pp, (long long)H * S * S, (long long)S * S, S,
        nullptr, nullptr, 0, 0,
        mask, (long long)S * S, S,
        1.0f / 32.0f, DH, H);
  }

  // ---- 4) softmax rows in place ----
  softmax_rows<<<dim3(B * H * S), blk, 0, stream>>>(Pp, S);

  // ---- 5) transpose v: vp [b*s][hd] -> vpT [b][hd][s] (aliases qp) ----
  {
    dim3 g(HD / 32, S / 32, B);
    bf16_T_batched<<<g, tblk, 0, stream>>>(vp, vpT, S, HD,
                                           (long long)S * HD, (long long)S * HD);
  }

  // ---- 6) ctx = P @ v -> bf16 [B*S][HD] ----
  {
    dim3 g(DH / 128, S / 128, B * H);
    gemm_bf16_wmma<false><<<g, blk, 0, stream>>>(
        Pp, (long long)H * S * S, (long long)S * S, S,
        vpT, (long long)HD * S, (long long)DH * S, S,
        ctx, (long long)S * HD, DH, HD,
        nullptr, nullptr, 0, 0, nullptr, 0, 0,
        1.0f, S, H);
  }

  // ---- 7) out = ctx @ Wo + bo + Q (residual), f32 into d_out ----
  {
    dim3 g(D / 128, (int)(BS / 128), 1);
    gemm_bf16_wmma<true><<<g, blk, 0, stream>>>(
        ctx, 0, 0, HD, WoT, 0, 0, HD,
        d_out, 0, 0, D,
        bo, Qin, 0, D,
        nullptr, 0, 0,
        1.0f, HD, 1);
  }

  // ---- 8) LayerNorm in place on d_out ----
  layernorm_rows<<<dim3((int)BS), blk, 0, stream>>>((float*)d_out, gamma, beta, D);
}
